// CNFBlock_82471962018033
// MI455X (gfx1250) — matrix operand order repaired and checked by reading, not compiled, not run on hardware
//
#include <hip/hip_runtime.h>
#include <hip/hip_bf16.h>
#include <math.h>
#include <stdint.h>

typedef __attribute__((ext_vector_type(16))) __bf16 v16bf;
typedef __attribute__((ext_vector_type(8)))  __bf16 v8bf;
typedef __attribute__((ext_vector_type(8)))  float  v8f;

#define GAS __attribute__((address_space(1)))   // global address space

#define EMB 256
#define NROWS 65536      // (SEQ*BATCH) * NUM_SAMPLED = 512*128
#define ODE_STEPS 8
#define DT (1.0f / 8.0f)
#define LOG2PI_TERM 235.2482644904f   // (EMB/2)*log(2*pi)

// ---------------------------------------------------------------------------
// prep 1: d[k] = sum_i W2[i,k]*Wx[k,i] ; wt[k] = wx_t[k] + wh_t[k]
// ---------------------------------------------------------------------------
__global__ void cnf_prep_small(const float* __restrict__ Wx,
                               const float* __restrict__ W2,
                               const float* __restrict__ wxt,
                               const float* __restrict__ wht,
                               float* __restrict__ d_out,
                               float* __restrict__ wt_out) {
  int k = blockIdx.x * blockDim.x + threadIdx.x;
  if (k < EMB) {
    float s = 0.f;
    for (int i = 0; i < EMB; ++i) s += W2[i * EMB + k] * Wx[k * EMB + i];
    d_out[k]  = s;
    wt_out[k] = wxt[k] + wht[k];
  }
}

// ---------------------------------------------------------------------------
// prep 2: ht[n][j] = h2[n] . Wh[j] + bx[j] + bh[j]   (512 x 256)
// ---------------------------------------------------------------------------
__global__ void cnf_prep_ht(const float* __restrict__ h2,
                            const float* __restrict__ Wh,
                            const float* __restrict__ bx,
                            const float* __restrict__ bh,
                            float* __restrict__ ht) {
  int n = blockIdx.x;          // 512
  int j = threadIdx.x;         // 256
  const float* hr = h2 + n * EMB;
  const float* wr = Wh + j * EMB;
  float s = 0.f;
  for (int c = 0; c < EMB; ++c) s += hr[c] * wr[c];
  ht[n * EMB + j] = s + bx[j] + bh[j];
}

// ---------------------------------------------------------------------------
// prep 3: pack Wx^T / W2^T into per-lane bf16 B-fragment order.
// B[k][n] = W[n][k].  Dense 16-bit B (32x16) layout (ISA sparse-B rule):
//   lane l holds N = l%16 ; h = l/16 ; element e of the 16-elem per-lane vector
//   -> K = 8h+e for e<8, K = 16+8h+(e-8) for e>=8 (matches A fragment).
// Storage: frag[((kt*16+nt)*32 + lane)*16 + e]
// ---------------------------------------------------------------------------
__global__ void cnf_prep_pack(const float* __restrict__ Wx,
                              const float* __restrict__ W2,
                              __bf16* __restrict__ WxB,
                              __bf16* __restrict__ W2B) {
  int b   = blockIdx.x;         // 256 = 2 mats * 8 kt * 16 nt
  int mat = b >> 7;
  int kt  = (b >> 4) & 7;
  int nt  = b & 15;
  int t    = threadIdx.x;       // 512 = 32 lanes * 16 elems
  int lane = t >> 4;
  int e    = t & 15;
  int Kl = ((lane < 16) ? 0 : 8) + ((e < 8) ? e : (e + 8));
  int Nl = lane & 15;
  const float* W = mat ? W2 : Wx;
  __bf16* dst    = mat ? W2B : WxB;
  float v = W[(nt * 16 + Nl) * EMB + (kt * 32 + Kl)];
  dst[((kt * 16 + nt) * 32 + lane) * 16 + e] = (__bf16)v;
}

// ---------------------------------------------------------------------------
// main: one wave (32 threads) per 16-row tile. 4096 blocks.
// ---------------------------------------------------------------------------
__global__ __launch_bounds__(32) void cnf_main(
    const float*  __restrict__ emb,
    const int*    __restrict__ tgts,
    const float*  __restrict__ h2,
    const float*  __restrict__ ht,
    const float*  __restrict__ wt,
    const float*  __restrict__ dvec,
    const float*  __restrict__ b2,
    const __bf16* __restrict__ WxB,
    const __bf16* __restrict__ W2B,
    float*        __restrict__ out) {
  __shared__ __align__(16) float  zbuf[2][16 * EMB];  // f32 z / z_acc ping-pong
  __shared__ __align__(16) __bf16 zs[16 * EMB];       // bf16 stage input / act
  __shared__ float cHT[EMB], cWT[EMB], cD[EMB], cB2[EMB];
  __shared__ float lpz[16];

  const int lane   = threadIdx.x;
  const int r0     = blockIdx.x * 16;
  const int n      = r0 >> 7;              // same n for whole tile (16 | 128)
  const int col    = lane & 15;
  const int rowOff = (lane < 16) ? 0 : 8;  // C-layout M offset for this lane

  // stage constants into LDS (keeps VGPR pressure low)
  for (int c = lane; c < EMB; c += 32) {
    cHT[c] = ht[n * EMB + c];
    cWT[c] = wt[c];
    cD[c]  = dvec[c];
    cB2[c] = b2[c];
  }

  // ---- gather z0 rows from emb_matrix; log p(z0) under N(h_n, I) ----
  const float* hrow = h2 + n * EMB;
  for (int m = 0; m < 16; ++m) {
    int tgt = tgts[r0 + m];
    const float* src = emb + (size_t)tgt * EMB;
    float part = 0.f;
    for (int c = lane; c < EMB; c += 32) {
      float v = src[c];
      zbuf[0][m * EMB + c] = v;
      zs[m * EMB + c]      = (__bf16)v;
      float df = v - hrow[c];
      part += df * df;
    }
    part += __shfl_xor(part, 16, 32);
    part += __shfl_xor(part, 8, 32);
    part += __shfl_xor(part, 4, 32);
    part += __shfl_xor(part, 2, 32);
    part += __shfl_xor(part, 1, 32);
    if (lane == 0) lpz[m] = -0.5f * part - LOG2PI_TERM;
  }
  __syncthreads();

  float tr[8];
#pragma unroll
  for (int j = 0; j < 8; ++j) tr[j] = 0.f;

  int cur = 0;
  uintptr_t wxb_p = (uintptr_t)WxB;
  uintptr_t w2b_p = (uintptr_t)W2B;
  const int aRowBase = (lane & 15) * EMB;       // A-frag source row in zs
  const int aKoff    = (lane < 16) ? 0 : 8;     // A-frag K offset for this lane

#pragma unroll 1
  for (int step = 0; step < ODE_STEPS; ++step) {
    const int nxt = cur ^ 1;
    // z_acc := z
    for (int idx = lane; idx < 16 * EMB; idx += 32) zbuf[nxt][idx] = zbuf[cur][idx];
    __syncthreads();
    const float tbase = (float)step * DT;

#pragma unroll 1
    for (int s = 0; s < 4; ++s) {
      const float toff = (s == 0) ? 0.f : ((s == 3) ? 1.f : 0.5f);
      const float ws   = (s == 0 || s == 3) ? (DT / 6.f) : (DT / 3.f);  // RK4 weight*dt
      const float cs   = (s < 2) ? (0.5f * DT) : DT;                    // next-stage step
      const float ts   = tbase + toff * DT;

      // ===================== GEMM1: pre = zs @ Wx^T =====================
      v16bf a[8];
#pragma unroll
      for (int kt = 0; kt < 8; ++kt) {
        const __bf16* ar = &zs[aRowBase + kt * 32 + aKoff];
        v8bf lo = *(const v8bf*)ar;
        v8bf hi = *(const v8bf*)(ar + 16);
        a[kt] = __builtin_shufflevector(lo, hi, 0, 1, 2, 3, 4, 5, 6, 7,
                                        8, 9, 10, 11, 12, 13, 14, 15);
      }
#pragma unroll
      for (int ntp = 0; ntp < 8; ++ntp) {
        asm volatile("" : "+s"(wxb_p));    // defeat LICM: keep B loads in-loop
        const GAS __bf16* wb = (const GAS __bf16*)wxb_p;   // pure global loads
        const int nt0 = 2 * ntp, nt1 = 2 * ntp + 1;
        v8f acc0 = (v8f){0, 0, 0, 0, 0, 0, 0, 0};
        v8f acc1 = (v8f){0, 0, 0, 0, 0, 0, 0, 0};
#pragma unroll
        for (int kt = 0; kt < 8; ++kt) {
          v16bf b0 = *(const GAS v16bf*)(wb + ((size_t)((kt * 16 + nt0) * 32 + lane)) * 16);
          v16bf b1 = *(const GAS v16bf*)(wb + ((size_t)((kt * 16 + nt1) * 32 + lane)) * 16);
          acc0 = __builtin_amdgcn_wmma_f32_16x16x32_bf16(false, a[kt], false, b0,
                                                         (short)0, acc0, false, false);
          acc1 = __builtin_amdgcn_wmma_f32_16x16x32_bf16(false, a[kt], false, b1,
                                                         (short)0, acc1, false, false);
        }
        // fused epilogue: bias + trace partial + softplus -> act (bf16 LDS)
#pragma unroll
        for (int half = 0; half < 2; ++half) {
          const int N = (half ? nt1 : nt0) * 16 + col;
          const v8f& ac = half ? acc1 : acc0;
          const float addn = cHT[N] + ts * cWT[N];
          const float ld   = cD[N];
#pragma unroll
          for (int j = 0; j < 8; ++j) {
            float pre = ac[j] + addn;
            float e   = __expf(-fabsf(pre));
            float sig = (pre >= 0.f) ? (1.f / (1.f + e)) : (e / (1.f + e));
            tr[j] += ws * sig * ld;
            float sp = fmaxf(pre, 0.f) + __logf(1.f + e);  // stable softplus
            zs[(j + rowOff) * EMB + N] = (__bf16)sp;
          }
        }
      }
      __syncthreads();

      // ===================== GEMM2: dz = act @ W2^T =====================
#pragma unroll
      for (int kt = 0; kt < 8; ++kt) {
        const __bf16* ar = &zs[aRowBase + kt * 32 + aKoff];
        v8bf lo = *(const v8bf*)ar;
        v8bf hi = *(const v8bf*)(ar + 16);
        a[kt] = __builtin_shufflevector(lo, hi, 0, 1, 2, 3, 4, 5, 6, 7,
                                        8, 9, 10, 11, 12, 13, 14, 15);
      }
      __syncthreads();   // act reads done before zs is overwritten below
#pragma unroll
      for (int ntp = 0; ntp < 8; ++ntp) {
        asm volatile("" : "+s"(w2b_p));
        const GAS __bf16* wb = (const GAS __bf16*)w2b_p;
        const int nt0 = 2 * ntp, nt1 = 2 * ntp + 1;
        v8f acc0 = (v8f){0, 0, 0, 0, 0, 0, 0, 0};
        v8f acc1 = (v8f){0, 0, 0, 0, 0, 0, 0, 0};
#pragma unroll
        for (int kt = 0; kt < 8; ++kt) {
          v16bf b0 = *(const GAS v16bf*)(wb + ((size_t)((kt * 16 + nt0) * 32 + lane)) * 16);
          v16bf b1 = *(const GAS v16bf*)(wb + ((size_t)((kt * 16 + nt1) * 32 + lane)) * 16);
          acc0 = __builtin_amdgcn_wmma_f32_16x16x32_bf16(false, a[kt], false, b0,
                                                         (short)0, acc0, false, false);
          acc1 = __builtin_amdgcn_wmma_f32_16x16x32_bf16(false, a[kt], false, b1,
                                                         (short)0, acc1, false, false);
        }
        // fused RK4 stage update: z_acc += ws*k ; zs = bf16(z + cs*k) (or new z)
#pragma unroll
        for (int half = 0; half < 2; ++half) {
          const int N = (half ? nt1 : nt0) * 16 + col;
          const v8f& ac = half ? acc1 : acc0;
          const float lb2 = cB2[N];
#pragma unroll
          for (int j = 0; j < 8; ++j) {
            int idx  = (j + rowOff) * EMB + N;
            float k  = ac[j] + lb2;
            float za = zbuf[nxt][idx] + ws * k;
            zbuf[nxt][idx] = za;
            float zn = (s < 3) ? (zbuf[cur][idx] + cs * k) : za;
            zs[idx] = (__bf16)zn;
          }
        }
      }
      __syncthreads();
    }
    cur ^= 1;
  }

  // ---- reduce trace partials within each 16-lane half; write output ----
#pragma unroll
  for (int j = 0; j < 8; ++j) {
    float v = tr[j];
    v += __shfl_xor(v, 1, 32);
    v += __shfl_xor(v, 2, 32);
    v += __shfl_xor(v, 4, 32);
    v += __shfl_xor(v, 8, 32);
    tr[j] = v;
  }
  if (lane == 0) {
#pragma unroll
    for (int j = 0; j < 8; ++j) out[r0 + j] = lpz[j] - tr[j];
  } else if (lane == 16) {
#pragma unroll
    for (int j = 0; j < 8; ++j) out[r0 + 8 + j] = lpz[8 + j] - tr[j];
  }
}

// ---------------------------------------------------------------------------
extern "C" void kernel_launch(void* const* d_in, const int* in_sizes, int n_in,
                              void* d_out, int out_size, void* d_ws, size_t ws_size,
                              hipStream_t stream) {
  (void)in_sizes; (void)n_in; (void)out_size; (void)ws_size;
  const float* h    = (const float*)d_in[0];   // (32,16,256) == h2 (512,256)
  const float* emb  = (const float*)d_in[1];   // (33278,256)
  const int*   tgt  = (const int*)  d_in[2];   // (512,128)
  const float* Wx   = (const float*)d_in[3];
  const float* wx_t = (const float*)d_in[4];
  const float* bx   = (const float*)d_in[5];
  const float* Wh   = (const float*)d_in[6];
  const float* wh_t = (const float*)d_in[7];
  const float* bh   = (const float*)d_in[8];
  const float* W2   = (const float*)d_in[9];
  const float* b2   = (const float*)d_in[10];
  float* out = (float*)d_out;

  // workspace layout (floats): d[256] | wt[256] | ht[512*256] | WxB | W2B
  float*  ws_f = (float*)d_ws;
  float*  d_d  = ws_f;
  float*  d_wt = ws_f + 256;
  float*  d_ht = ws_f + 512;
  __bf16* WxB  = (__bf16*)(ws_f + 512 + 512 * 256);
  __bf16* W2B  = WxB + 8 * 16 * 32 * 16;

  cnf_prep_small<<<1, 256, 0, stream>>>(Wx, W2, wx_t, wh_t, d_d, d_wt);
  cnf_prep_ht<<<512, 256, 0, stream>>>(h, Wh, bx, bh, d_ht);
  cnf_prep_pack<<<256, 512, 0, stream>>>(Wx, W2, WxB, W2B);
  cnf_main<<<NROWS / 16, 32, 0, stream>>>(emb, tgt, h, d_ht, d_wt, d_d, b2,
                                          WxB, W2B, out);
}
